// SparseSegHead_71305047048340
// MI455X (gfx1250) — compile-verified, matrix operand order checked by reading, not run on hardware
//
#include <hip/hip_runtime.h>

typedef __attribute__((ext_vector_type(16))) _Float16 v16h;
typedef __attribute__((ext_vector_type(8)))  float    v8f;

constexpr int C_IN  = 32;
constexpr int H_CH  = 64;
constexpr int OUT_CH = 3;
constexpr int DIM   = 128;
constexpr long long VOX = (long long)DIM * DIM * DIM;   // 2,097,152 voxels per channel

// One wave32 handles a tile of 16 points:
//   layer 1: hidden[16x64] = relu(A[16x32] * W1^T[32x64] + b1)  via 4x v_wmma_f32_16x16x32_f16
//   layer 2: seg[16x3]     = hidden * W2^T + b2                 via VALU + shfl_xor reduce
__global__ __launch_bounds__(256) void sparse_seg_head_kernel(
    const float* __restrict__ hvol,     // [32, 128,128,128]
    const float* __restrict__ points,   // [N, 3]
    const float* __restrict__ W1,       // [64, 32]
    const float* __restrict__ b1,       // [64]
    const float* __restrict__ W2,       // [3, 64]
    const float* __restrict__ b2,       // [3]
    float* __restrict__ out,            // [N, 3]
    int n_points)
{
    const int lane       = threadIdx.x & 31;
    const int waveInBlk  = threadIdx.x >> 5;
    const int waveGlobal = blockIdx.x * (blockDim.x >> 5) + waveInBlk;
    const int tileBase   = waveGlobal * 16;
    if (tileBase >= n_points) return;            // wave-uniform

    const int row  = lane & 15;                  // matrix row (point within tile)
    const int half = lane >> 4;                  // 0: K lo-groups, 1: K hi-groups

    // ---- flat voxel index, exactly like the reference ----
    int p = tileBase + row;
    if (p > n_points - 1) p = n_points - 1;      // clamp (no EXEC divergence)
    const float px = points[p * 3 + 0];
    const float py = points[p * 3 + 1];
    const float pz = points[p * 3 + 2];
    const float find = floorf(px) * (float)(DIM * DIM) + floorf(py) * (float)DIM + floorf(pz);
    const long long ind = (long long)(int)find;

    // ---- gather A (16 points x 32 chan) straight into the f16 A layout ----
    // lanes 0-15 : M=lane,    K = {0..7, 16..23}
    // lanes 16-31: M=lane-16, K = {8..15, 24..31}
    v16h a;
#pragma unroll
    for (int j = 0; j < 16; ++j) {
        const int c = (j < 8) ? (half * 8 + j) : (16 + half * 8 + (j - 8));
        a[j] = (_Float16)hvol[(long long)c * VOX + ind];
    }

    // ---- B tiles of W1 (K=32 x N=64 -> 4 tiles), mirrored packing, lane = column ----
    const int colLocal = lane & 15;
    v16h btile[4];
#pragma unroll
    for (int t = 0; t < 4; ++t) {
        const int n = t * 16 + colLocal;         // hidden channel (B column)
#pragma unroll
        for (int j = 0; j < 16; ++j) {
            const int c = (j < 8) ? (half * 8 + j) : (16 + half * 8 + (j - 8));
            btile[t][j] = (_Float16)W1[n * C_IN + c];
        }
    }

    // ---- layer 1 GEMM: 4x WMMA ----
    v8f acc[4];
#pragma unroll
    for (int t = 0; t < 4; ++t) {
        v8f c0 = {};
        acc[t] = __builtin_amdgcn_wmma_f32_16x16x32_f16(
            /*neg_a=*/false, a, /*neg_b=*/false, btile[t],
            /*c_mod=*/(short)0, c0, /*reuse_a=*/false, /*reuse_b=*/false);
    }

    // ---- bias + relu; D layout: VGPR r -> M = r + 8*half, column = lane&15 ----
    float hd[4][8];
#pragma unroll
    for (int t = 0; t < 4; ++t) {
        const float bias = b1[t * 16 + colLocal];
#pragma unroll
        for (int r = 0; r < 8; ++r) {
            const float v = acc[t][r] + bias;
            hd[t][r] = v > 0.f ? v : 0.f;
        }
    }

    // ---- layer 2: per-lane partial dot over its hidden channel, reduce in-half ----
    float w2v[4][OUT_CH];
#pragma unroll
    for (int t = 0; t < 4; ++t)
#pragma unroll
        for (int o = 0; o < OUT_CH; ++o)
            w2v[t][o] = W2[o * H_CH + t * 16 + colLocal];

    float part[8][OUT_CH];
#pragma unroll
    for (int r = 0; r < 8; ++r)
#pragma unroll
        for (int o = 0; o < OUT_CH; ++o) {
            float s = 0.f;
#pragma unroll
            for (int t = 0; t < 4; ++t) s += hd[t][r] * w2v[t][o];
            part[r][o] = s;
        }

    // xor masks 1..8 only flip low 4 lane bits -> reduction stays inside each 16-lane half
#pragma unroll
    for (int mask = 1; mask <= 8; mask <<= 1)
#pragma unroll
        for (int r = 0; r < 8; ++r)
#pragma unroll
            for (int o = 0; o < OUT_CH; ++o)
                part[r][o] += __shfl_xor(part[r][o], mask, 32);

    // lane 0 of each half writes its 8 rows
    if (colLocal == 0) {
#pragma unroll
        for (int r = 0; r < 8; ++r) {
            const int m = tileBase + half * 8 + r;
            if (m < n_points) {
#pragma unroll
                for (int o = 0; o < OUT_CH; ++o)
                    out[m * OUT_CH + o] = part[r][o] + b2[o];
            }
        }
    }
}

extern "C" void kernel_launch(void* const* d_in, const int* in_sizes, int n_in,
                              void* d_out, int out_size, void* d_ws, size_t ws_size,
                              hipStream_t stream) {
    const float* hvol   = (const float*)d_in[0];
    const float* points = (const float*)d_in[1];
    const float* W1     = (const float*)d_in[2];
    const float* b1     = (const float*)d_in[3];
    const float* W2     = (const float*)d_in[4];
    const float* b2     = (const float*)d_in[5];
    float* out = (float*)d_out;

    const int n_points = in_sizes[1] / 3;                 // points is [1, N, 3]
    const int tiles    = (n_points + 15) / 16;            // 16 points per wave
    const int wavesPerBlock = 8;                          // 256 threads
    const int blocks   = (tiles + wavesPerBlock - 1) / wavesPerBlock;

    sparse_seg_head_kernel<<<blocks, wavesPerBlock * 32, 0, stream>>>(
        hvol, points, W1, b1, W2, b2, out, n_points);
}
